// KNNModel_58763742544380
// MI455X (gfx1250) — matrix operand order repaired and checked by reading, not compiled, not run on hardware
//
#include <hip/hip_runtime.h>
#include <hip/hip_bf16.h>
#include <math.h>

typedef __attribute__((ext_vector_type(16))) _Float16 v16h;
typedef __attribute__((ext_vector_type(8)))  float    v8f;
typedef __attribute__((ext_vector_type(4)))  float    f4;

#define B_ROWS 512
#define DIN    512
#define DDIM   256
#define NTRAIN 500000
#define CCLS   1000
#define KTOP   7
#define TEMPR  0.07f
#define COLS_PER_BLOCK 512
#define NBLKN  ((NTRAIN + COLS_PER_BLOCK - 1) / COLS_PER_BLOCK)   /* 977 */
#define CAND_PER_ROW (NBLKN * KTOP)

/* workspace layout (bytes) */
#define ZFRAG_BYTES ((size_t)B_ROWS * DDIM * 2)                   /* 262144   */
#define CANDV_OFF   ZFRAG_BYTES
#define CANDV_BYTES ((size_t)B_ROWS * CAND_PER_ROW * 4)           /* ~14.0 MB */
#define CANDI_OFF   (CANDV_OFF + CANDV_BYTES)

/* LDS layout for sims kernel */
#define ZS_BYTES    (B_ROWS * DDIM * 2)     /* 262144 */
#define STAGE_BYTES (8 * 4 * 16 * 16 * 4)   /* 8 waves * 4 rowtiles * 16x16 f32 = 32768 */
#define SIMS_LDS    (ZS_BYTES + STAGE_BYTES)

__device__ __forceinline__ void topk_insert(float (&tv)[KTOP], int (&ti)[KTOP],
                                            float v, int id) {
  /* tv kept sorted descending; tv[6] is current min. Guard keeps this cheap
     (EXEC-masked branch, rarely taken once the heap warms up). */
  if (id < NTRAIN && v > tv[KTOP - 1]) {
    float nv = v; int ni = id;
#pragma unroll
    for (int q = 0; q < KTOP; ++q) {
      bool g = nv > tv[q];
      float ov = tv[q]; int oi = ti[q];
      tv[q] = g ? nv : ov;  ti[q] = g ? ni : oi;
      nv   = g ? ov : nv;   ni   = g ? oi : ni;
    }
  }
}

/* ------------------------------------------------------------------ */
/* Kernel 1: preprocess one row -> normalized z -> f16 A-fragment img  */
/* A-fragment (16-bit, 16x32, §7.12.2): lane = m + 16*khalf,           */
/*   VGPR j holds K = (j<4 ? 2j : 16+2(j-4)) + 8*khalf, pairs (K,K+1). */
/* Fragment (rt,kb) = 1024B at ((rt*8+kb)*1024), lane-contiguous 32B.  */
/* ------------------------------------------------------------------ */
__global__ __launch_bounds__(256) void prep_kernel(
    const float* __restrict__ x, const float* __restrict__ mean1,
    const float* __restrict__ std1, const float* __restrict__ P,
    const float* __restrict__ mean2, const float* __restrict__ std2,
    _Float16* __restrict__ zfrag) {
  __shared__ float xs[DIN];
  __shared__ float red[256];
  const int row = blockIdx.x;
  const int t = threadIdx.x;
  for (int i = t; i < DIN; i += 256)
    xs[i] = (x[row * DIN + i] - mean1[i]) / std1[i];
  __syncthreads();
  float acc = 0.f;
  for (int k = 0; k < DIN; ++k) acc += xs[k] * P[k * DDIM + t];
  acc = (acc - mean2[t]) / std2[t];
  red[t] = acc * acc;
  __syncthreads();
  for (int s = 128; s > 0; s >>= 1) {
    if (t < s) red[t] += red[t + s];
    __syncthreads();
  }
  const float norm = sqrtf(red[0]);
  const float zv = acc / fmaxf(norm, 1e-12f);
  /* write into A-fragment layout; this thread owns column kk = t of row */
  const int rt = row >> 4, m = row & 15;
  const int kb = t >> 5, K = t & 31;
  const int khalf = (K >> 3) & 1;
  const int Kp = K - 8 * khalf;                       /* {0..7} u {16..23} */
  const int j = (Kp < 8) ? (Kp >> 1) : (4 + ((Kp - 16) >> 1));
  const int h = 2 * j + (K & 1);
  const int lane = m + 16 * khalf;
  const size_t off = (size_t)((rt * 8 + kb) * 1024) + (size_t)lane * 32 + (size_t)h * 2;
  *(_Float16*)((char*)zfrag + off) = (_Float16)zv;
}

/* ------------------------------------------------------------------ */
/* Kernel 2: fused sims GEMM (f16 WMMA) + per-row running top-7        */
/* Block = 256 threads (8 waves). Wave w owns rows [64w, 64w+64).      */
/* All waves share the same 16-column tile; train_feats read once.     */
/* ------------------------------------------------------------------ */
__global__ __launch_bounds__(256) void sims_topk_kernel(
    const _Float16* __restrict__ zfrag, const float* __restrict__ tf,
    float* __restrict__ candV, int* __restrict__ candI) {
  extern __shared__ char smem[];
  _Float16* zs   = (_Float16*)smem;                 /* 256 KB A fragments  */
  float*    stage = (float*)(smem + ZS_BYTES);      /* 32 KB C staging     */

  const int t = threadIdx.x;
  const int w = t >> 5;           /* wave id 0..7  */
  const int l = t & 31;           /* lane  0..31   */
  const int blk = blockIdx.x;

  /* stage z fragments into LDS once (reads hit L2: z is only 256 KB) */
  {
    const f4* src = (const f4*)zfrag;
    f4* dst = (f4*)zs;
    for (int i = t; i < (int)(ZS_BYTES / 16); i += 256) dst[i] = src[i];
  }
  __syncthreads();

  /* per-lane top-7 state for 2 rows this lane owns */
  float tv0[KTOP], tv1[KTOP]; int ti0[KTOP], ti1[KTOP];
#pragma unroll
  for (int k = 0; k < KTOP; ++k) {
    tv0[k] = -INFINITY; tv1[k] = -INFINITY; ti0[k] = 0; ti1[k] = 0;
  }

  const int n_lane = l & 15;          /* B column within tile  */
  const int halfsel = l >> 4;         /* B K-half select       */
  float* mystage = stage + w * 1024;  /* 4 rowtiles * 256 f32  */
  const int lr0 = 2 * l, lr1 = 2 * l + 1;
  const float* srow0 = mystage + (lr0 >> 4) * 256 + (lr0 & 15) * 16;
  const float* srow1 = mystage + (lr1 >> 4) * 256 + (lr1 & 15) * 16;
  const int colbase0 = blk * COLS_PER_BLOCK;

  for (int it = 0; it < COLS_PER_BLOCK / 16; ++it) {
    const int colbase = colbase0 + it * 16;
    int mycol = colbase + n_lane;
    int ccol = (mycol < NTRAIN) ? mycol : (NTRAIN - 1);

    /* ---- load B fragments for all 8 K-blocks, convert f32->f16 ---- */
    /* B (32x16) lane map: col n = l&15, elems 0..7 hold K = 8*half + s,
       elems 8..15 hold K = 16 + 8*half + (s-8) -> two 8-float runs per kb */
    v16h Bf[8];
    const float* bp = tf + (size_t)ccol * DDIM + halfsel * 8;
#pragma unroll
    for (int kb = 0; kb < 8; ++kb) {
      const f4* p = (const f4*)(bp + kb * 32);
      f4 a = p[0], b = p[1], c = p[4], d = p[5];
      v16h bf;
      bf[0]  = (_Float16)a.x;  bf[1]  = (_Float16)a.y;
      bf[2]  = (_Float16)a.z;  bf[3]  = (_Float16)a.w;
      bf[4]  = (_Float16)b.x;  bf[5]  = (_Float16)b.y;
      bf[6]  = (_Float16)b.z;  bf[7]  = (_Float16)b.w;
      bf[8]  = (_Float16)c.x;  bf[9]  = (_Float16)c.y;
      bf[10] = (_Float16)c.z;  bf[11] = (_Float16)c.w;
      bf[12] = (_Float16)d.x;  bf[13] = (_Float16)d.y;
      bf[14] = (_Float16)d.z;  bf[15] = (_Float16)d.w;
      Bf[kb] = bf;
    }
    /* prefetch next tile's column (gfx1250 global_prefetch_b8) */
    {
      int pcol = mycol + 16; if (pcol >= NTRAIN) pcol = NTRAIN - 1;
      __builtin_prefetch(tf + (size_t)pcol * DDIM + halfsel * 16, 0, 1);
    }

    /* ---- WMMA: 4 rowtiles x 8 kblocks, K=256 ---- */
    v8f Cc[4];
#pragma unroll
    for (int rt = 0; rt < 4; ++rt) {
      v8f z8 = {0.f, 0.f, 0.f, 0.f, 0.f, 0.f, 0.f, 0.f};
      Cc[rt] = z8;
    }
#pragma unroll
    for (int rt = 0; rt < 4; ++rt) {
#pragma unroll
      for (int kb = 0; kb < 8; ++kb) {
        v16h Af = ((const v16h*)zs)[((w * 4 + rt) * 8 + kb) * 32 + l];
        Cc[rt] = __builtin_amdgcn_wmma_f32_16x16x32_f16(
            false, Af, false, Bf[kb], (short)0, Cc[rt], false, false);
      }
    }

    /* ---- spill C tiles to wave-private LDS staging (row-major) ---- */
#pragma unroll
    for (int rt = 0; rt < 4; ++rt) {
#pragma unroll
      for (int v = 0; v < 8; ++v) {
        mystage[rt * 256 + (v + 8 * halfsel) * 16 + n_lane] = Cc[rt][v];
      }
    }
    /* same-wave LDS ops are in-order; no cross-wave sharing -> no barrier */

    /* ---- top-7 update: this lane scans 2 full rows x 16 cols ---- */
#pragma unroll
    for (int n4 = 0; n4 < 4; ++n4) {
      const int base = colbase + n4 * 4;
      f4 va = *(const f4*)(srow0 + n4 * 4);
      topk_insert(tv0, ti0, va.x, base + 0);
      topk_insert(tv0, ti0, va.y, base + 1);
      topk_insert(tv0, ti0, va.z, base + 2);
      topk_insert(tv0, ti0, va.w, base + 3);
      f4 vb = *(const f4*)(srow1 + n4 * 4);
      topk_insert(tv1, ti1, vb.x, base + 0);
      topk_insert(tv1, ti1, vb.y, base + 1);
      topk_insert(tv1, ti1, vb.z, base + 2);
      topk_insert(tv1, ti1, vb.w, base + 3);
    }
  }

  /* ---- emit this block's per-row candidates ---- */
  {
    const int row0 = w * 64 + lr0;
    const int row1 = w * 64 + lr1;
    size_t b0 = ((size_t)row0 * NBLKN + blk) * KTOP;
    size_t b1 = ((size_t)row1 * NBLKN + blk) * KTOP;
#pragma unroll
    for (int k = 0; k < KTOP; ++k) { candV[b0 + k] = tv0[k]; candI[b0 + k] = ti0[k]; }
#pragma unroll
    for (int k = 0; k < KTOP; ++k) { candV[b1 + k] = tv1[k]; candI[b1 + k] = ti1[k]; }
  }
}

/* ------------------------------------------------------------------ */
/* Kernel 3: per-row candidate merge -> exp weights -> class scatter   */
/* ------------------------------------------------------------------ */
__global__ __launch_bounds__(256) void merge_out_kernel(
    const float* __restrict__ candV, const int* __restrict__ candI,
    const int* __restrict__ labels, float* __restrict__ out) {
  __shared__ float sv[256 * KTOP];
  __shared__ int   si[256 * KTOP];
  const int row = blockIdx.x;
  const int t = threadIdx.x;
  const float* cv = candV + (size_t)row * CAND_PER_ROW;
  const int*   ci = candI + (size_t)row * CAND_PER_ROW;

  float tv[KTOP]; int ti[KTOP];
#pragma unroll
  for (int k = 0; k < KTOP; ++k) { tv[k] = -INFINITY; ti[k] = 0; }
  for (int i = t; i < CAND_PER_ROW; i += 256)
    topk_insert(tv, ti, cv[i], ci[i]);
#pragma unroll
  for (int k = 0; k < KTOP; ++k) { sv[t * KTOP + k] = tv[k]; si[t * KTOP + k] = ti[k]; }
  __syncthreads();

  float* orow = out + (size_t)row * CCLS;
  for (int i = t; i < CCLS; i += 256) orow[i] = 0.f;
  __syncthreads();

  if (t == 0) {
    float fv[KTOP]; int fi[KTOP];
#pragma unroll
    for (int k = 0; k < KTOP; ++k) { fv[k] = -INFINITY; fi[k] = 0; }
    for (int i = 0; i < 256 * KTOP; ++i)
      topk_insert(fv, fi, sv[i], si[i]);
    /* fv is sorted descending: fv[0] is the row max */
    const float v0 = fv[0];
#pragma unroll
    for (int k = 0; k < KTOP; ++k) {
      float wgt = expf((fv[k] - v0) / TEMPR);
      orow[labels[fi[k]]] += wgt;   /* serial: duplicate labels accumulate */
    }
  }
}

/* ------------------------------------------------------------------ */
extern "C" void kernel_launch(void* const* d_in, const int* in_sizes, int n_in,
                              void* d_out, int out_size, void* d_ws, size_t ws_size,
                              hipStream_t stream) {
  const float* x      = (const float*)d_in[0];
  const float* mean1  = (const float*)d_in[1];
  const float* std1   = (const float*)d_in[2];
  const float* P      = (const float*)d_in[3];
  const float* mean2  = (const float*)d_in[4];
  const float* std2   = (const float*)d_in[5];
  const float* tfeats = (const float*)d_in[6];
  const int*   tlbl   = (const int*)d_in[7];
  float* out = (float*)d_out;

  _Float16* zfrag = (_Float16*)d_ws;
  float*    candV = (float*)((char*)d_ws + CANDV_OFF);
  int*      candI = (int*)((char*)d_ws + CANDI_OFF);

  (void)in_sizes; (void)n_in; (void)out_size; (void)ws_size;

  hipFuncSetAttribute((const void*)sims_topk_kernel,
                      hipFuncAttributeMaxDynamicSharedMemorySize, SIMS_LDS);

  prep_kernel<<<B_ROWS, 256, 0, stream>>>(x, mean1, std1, P, mean2, std2, zfrag);
  sims_topk_kernel<<<NBLKN, 256, SIMS_LDS, stream>>>(zfrag, tfeats, candV, candI);
  merge_out_kernel<<<B_ROWS, 256, 0, stream>>>(candV, candI, tlbl, out);
}
